// MegaCRN_73770358276530
// MI455X (gfx1250) — matrix operand order repaired and checked by reading, not compiled, not run on hardware
//
#include <hip/hip_runtime.h>
#include <hip/hip_bf16.h>

typedef __bf16 bf16_t;
typedef __attribute__((ext_vector_type(2)))  __bf16 v2bf;
typedef __attribute__((ext_vector_type(16))) __bf16 v16bf;
typedef __attribute__((ext_vector_type(8)))  float  v8f;

#define BMT 128
#define BNT 128
#define BKT 32
#define LPAD 8

// ============ bf16 WMMA GEMM, async-LDS staging (edge-free diffusion path) ============
// REQUIRES: M%128==0, N%128==0, K%32==0, lda/ldb/ldc %8==0, 16B-aligned bases.
// Cb = A@B (bf16 out).
__global__ __launch_bounds__(256)
void gemm_bb_async(const bf16_t* __restrict__ A, int lda,
                   const bf16_t* __restrict__ B, int ldb,
                   bf16_t* __restrict__ Cb, int ldc,
                   int M, int N, int K)
{
    __shared__ bf16_t shA[2][BMT][BKT + LPAD];
    __shared__ bf16_t shB[2][BKT][BNT + LPAD];

    const int m0 = blockIdx.y * BMT;
    const int n0 = blockIdx.x * BNT;
    const int tid  = threadIdx.x;
    const int lane = tid & 31;
    const int wave = tid >> 5;
    const int wr   = wave & 3;
    const int wc   = wave >> 2;

    const int ar0 = tid >> 2;           // A row chunk0 (0..63), chunk1 = +64
    const int ac  = (tid & 3) << 3;     // A col (8 bf16 = 16B chunks)
    const int br0 = tid >> 4;           // B row chunk0 (0..15), chunk1 = +16
    const int bc  = (tid & 15) << 3;    // B col

    // issue 4 async 16B copies per thread (ASYNCcnt += 4 per wave)
    auto issue = [&](int k0, int buf) {
        #pragma unroll
        for (int i = 0; i < 2; ++i) {
            unsigned long long g = (unsigned long long)(uintptr_t)
                (A + (long long)(m0 + ar0 + i*64) * lda + (k0 + ac));
            unsigned l = (unsigned)(uintptr_t)&shA[buf][ar0 + i*64][ac];
            asm volatile("global_load_async_to_lds_b128 %0, %1, off"
                         :: "v"(l), "v"(g) : "memory");
        }
        #pragma unroll
        for (int i = 0; i < 2; ++i) {
            unsigned long long g = (unsigned long long)(uintptr_t)
                (B + (long long)(k0 + br0 + i*16) * ldb + (n0 + bc));
            unsigned l = (unsigned)(uintptr_t)&shB[buf][br0 + i*16][bc];
            asm volatile("global_load_async_to_lds_b128 %0, %1, off"
                         :: "v"(l), "v"(g) : "memory");
        }
    };

    v8f acc[2][4];
    const v8f zero8 = {0.f,0.f,0.f,0.f,0.f,0.f,0.f,0.f};
    #pragma unroll
    for (int s = 0; s < 2; ++s)
        #pragma unroll
        for (int t = 0; t < 4; ++t) acc[s][t] = zero8;

    issue(0, 0);
    int cur = 0;
    for (int k0 = 0; ; ) {
        const int kn  = k0 + BKT;
        const bool more = kn < K;
        __syncthreads();                      // prior readers of buf cur^1 are done
        if (more) {
            issue(kn, cur ^ 1);               // next tile copies overlap WMMAs below
            asm volatile("s_wait_asynccnt 0x4" ::: "memory");  // current tile resident
        } else {
            asm volatile("s_wait_asynccnt 0x0" ::: "memory");
        }
        __syncthreads();                      // cross-wave visibility of LDS writes

        v16bf afr[2];
        #pragma unroll
        for (int s = 0; s < 2; ++s) {
            int m  = wr*32 + s*16 + (lane & 15);
            int kh = (lane >> 4) * 8;
            #pragma unroll
            for (int r = 0; r < 8; ++r) {
                int k = ((r < 4) ? 0 : 16) + kh + (r & 3)*2;
                afr[s][2*r]   = shA[cur][m][k];
                afr[s][2*r+1] = shA[cur][m][k+1];
            }
        }
        v16bf bfr[4];
        #pragma unroll
        for (int t = 0; t < 4; ++t) {
            int nb = wc*64 + t*16;
            #pragma unroll
            for (int r = 0; r < 8; ++r) {
                int k = ((r < 4) ? 0 : 16) + (lane >> 4) * 8 + (r & 3)*2;
                bfr[t][2*r]   = shB[cur][k][nb + (lane & 15)];
                bfr[t][2*r+1] = shB[cur][k+1][nb + (lane & 15)];
            }
        }
        #pragma unroll
        for (int s = 0; s < 2; ++s)
            #pragma unroll
            for (int t = 0; t < 4; ++t)
                acc[s][t] = __builtin_amdgcn_wmma_f32_16x16x32_bf16(
                    false, afr[s], false, bfr[t], (short)0, acc[s][t], false, false);

        if (!more) break;
        cur ^= 1;
        k0 = kn;
    }

    #pragma unroll
    for (int s = 0; s < 2; ++s) {
        #pragma unroll
        for (int t = 0; t < 4; ++t) {
            int n = n0 + wc*64 + t*16 + (lane & 15);
            #pragma unroll
            for (int v = 0; v < 8; ++v) {
                int m = m0 + wr*32 + s*16 + v + 8*(lane >> 4);
                Cb[(long long)m*ldc + n] = (bf16_t)acc[s][t][v];
            }
        }
    }
}

// ================= bf16-native WMMA GEMM, register staging (linear GEMMs) =================
__global__ __launch_bounds__(256)
void gemm_bb_wmma(const bf16_t* __restrict__ A, int lda,
                  const bf16_t* __restrict__ B, int ldb,
                  float* __restrict__ Cf, bf16_t* __restrict__ Cb, int ldc,
                  const float* __restrict__ bias,
                  int M, int N, int K, int accum, int act)
{
    __shared__ bf16_t shA[BMT][BKT + LPAD];
    __shared__ bf16_t shB[BKT][BNT + LPAD];

    const int m0 = blockIdx.y * BMT;
    const int n0 = blockIdx.x * BNT;
    const int tid  = threadIdx.x;
    const int lane = tid & 31;
    const int wave = tid >> 5;
    const int wr   = wave & 3;
    const int wc   = wave >> 2;

    const int M1 = M - 1, N1 = N - 1, K1 = K - 1;
    const bool edgeM = (m0 + BMT > M);
    const bool edgeN = (n0 + BNT > N);

    const int ar0 = tid >> 2;
    const int ac  = (tid & 3) << 3;
    const int br0 = tid >> 4;
    const int bc  = (tid & 15) << 3;

    union Ch { uint4 u; bf16_t h[8]; };
    Ch ach[2], bch[2];
    int kld = 0;

    auto load_tiles = [&](int k0) {
        kld = k0;
        const bool ae = edgeM || (k0 + BKT > K);
        if (!ae) {
            #pragma unroll
            for (int i = 0; i < 2; ++i)
                ach[i].u = *(const uint4*)(A + (long long)(m0 + ar0 + i*64) * lda + (k0 + ac));
        } else {
            #pragma unroll
            for (int i = 0; i < 2; ++i) {
                int gm = m0 + ar0 + i*64;
                int gmc = gm < M1 ? gm : M1;
                const bf16_t* ap = A + (long long)gmc * lda;
                #pragma unroll
                for (int j = 0; j < 8; ++j) {
                    int gk = k0 + ac + j;
                    ach[i].h[j] = ap[gk < K1 ? gk : K1];
                }
            }
        }
        const bool be = edgeN || (k0 + BKT > K);
        if (!be) {
            #pragma unroll
            for (int i = 0; i < 2; ++i)
                bch[i].u = *(const uint4*)(B + (long long)(k0 + br0 + i*16) * ldb + (n0 + bc));
        } else {
            #pragma unroll
            for (int i = 0; i < 2; ++i) {
                int gk = k0 + br0 + i*16;
                const bf16_t* bp = B + (long long)(gk < K1 ? gk : K1) * ldb;
                #pragma unroll
                for (int j = 0; j < 8; ++j) {
                    int gn = n0 + bc + j;
                    bch[i].h[j] = bp[gn < N1 ? gn : N1];
                }
            }
        }
    };
    auto store_tiles = [&]() {
        const bool ae = edgeM || (kld + BKT > K);
        #pragma unroll
        for (int i = 0; i < 2; ++i) {
            if (ae) {
                int gm = m0 + ar0 + i*64;
                #pragma unroll
                for (int j = 0; j < 8; ++j) {
                    int gk = kld + ac + j;
                    if (!(gm < M && gk < K)) ach[i].h[j] = (bf16_t)0.0f;
                }
            }
            *(uint4*)&shA[ar0 + i*64][ac] = ach[i].u;
        }
        const bool be = edgeN || (kld + BKT > K);
        #pragma unroll
        for (int i = 0; i < 2; ++i) {
            if (be) {
                int gk = kld + br0 + i*16;
                #pragma unroll
                for (int j = 0; j < 8; ++j) {
                    int gn = n0 + bc + j;
                    if (!(gk < K && gn < N)) bch[i].h[j] = (bf16_t)0.0f;
                }
            }
            *(uint4*)&shB[br0 + i*16][bc] = bch[i].u;
        }
    };

    v8f acc[2][4];
    const v8f zero8 = {0.f,0.f,0.f,0.f,0.f,0.f,0.f,0.f};
    #pragma unroll
    for (int s = 0; s < 2; ++s)
        #pragma unroll
        for (int t = 0; t < 4; ++t) acc[s][t] = zero8;

    load_tiles(0);
    store_tiles();
    __syncthreads();

    for (int k0 = 0; ; ) {
        const int kn  = k0 + BKT;
        const bool more = kn < K;
        if (more) load_tiles(kn);

        v16bf afr[2];
        #pragma unroll
        for (int s = 0; s < 2; ++s) {
            int m  = wr*32 + s*16 + (lane & 15);
            int kh = (lane >> 4) * 8;
            #pragma unroll
            for (int r = 0; r < 8; ++r) {
                int k = ((r < 4) ? 0 : 16) + kh + (r & 3)*2;
                afr[s][2*r]   = shA[m][k];
                afr[s][2*r+1] = shA[m][k+1];
            }
        }
        v16bf bfr[4];
        #pragma unroll
        for (int t = 0; t < 4; ++t) {
            int nb = wc*64 + t*16;
            #pragma unroll
            for (int r = 0; r < 8; ++r) {
                int k = ((r < 4) ? 0 : 16) + (lane >> 4) * 8 + (r & 3)*2;
                bfr[t][2*r]   = shB[k][nb + (lane & 15)];
                bfr[t][2*r+1] = shB[k+1][nb + (lane & 15)];
            }
        }
        #pragma unroll
        for (int s = 0; s < 2; ++s)
            #pragma unroll
            for (int t = 0; t < 4; ++t)
                acc[s][t] = __builtin_amdgcn_wmma_f32_16x16x32_bf16(
                    false, afr[s], false, bfr[t], (short)0, acc[s][t], false, false);

        __syncthreads();
        if (!more) break;
        store_tiles();
        __syncthreads();
        k0 = kn;
    }

    #pragma unroll
    for (int s = 0; s < 2; ++s) {
        #pragma unroll
        for (int t = 0; t < 4; ++t) {
            int n = n0 + wc*64 + t*16 + (lane & 15);
            #pragma unroll
            for (int v = 0; v < 8; ++v) {
                int m = m0 + wr*32 + s*16 + v + 8*(lane >> 4);
                if (m < M && n < N) {
                    float val = acc[s][t][v];
                    if (accum) val += Cf[(long long)m*ldc + n];
                    if (bias)  val += bias[n];
                    if      (act == 1) val = fmaxf(val, 0.0f);
                    else if (act == 2) val = 1.0f / (1.0f + __expf(-val));
                    else if (act == 3) val = tanhf(val);
                    if (Cb) Cb[(long long)m*ldc + n] = (bf16_t)val;
                    else    Cf[(long long)m*ldc + n] = val;
                }
            }
        }
    }
}

// ================= f32-staging WMMA GEMM (cheap one-off GEMMs) =================
template <int TRANSB>
__global__ __launch_bounds__(256)
void gemm_f32_wmma(const float* __restrict__ A, int lda,
                   const float* __restrict__ B, int ldb,
                   float* __restrict__ C, int ldc,
                   const float* __restrict__ bias,
                   int M, int N, int K, float alpha, int accum, int act)
{
    __shared__ bf16_t shA[BMT][BKT + LPAD];
    __shared__ bf16_t shB[BKT][BNT + LPAD];

    const int m0 = blockIdx.y * BMT;
    const int n0 = blockIdx.x * BNT;
    const int tid  = threadIdx.x;
    const int lane = tid & 31;
    const int wave = tid >> 5;
    const int wr   = wave & 3;
    const int wc   = wave >> 2;

    const int M1 = M - 1, N1 = N - 1, K1 = K - 1;
    const bool edgeM = (m0 + BMT > M);
    const bool edgeN = (n0 + BNT > N);

    const int arow = tid >> 4;
    const int acol = (tid & 15) << 1;
    const int brow = tid >> 6;
    const int bcol = (tid & 63) << 1;

    float are[16], bre[16];
    int kld = 0;

    auto load_tiles = [&](int k0) {
        kld = k0;
        const bool ae = edgeM || (k0 + BKT > K);
        if (!ae) {
            #pragma unroll
            for (int i = 0; i < 8; ++i) {
                const float* ap = A + (long long)(m0 + arow + i*16) * lda + (k0 + acol);
                are[2*i]   = ap[0];
                are[2*i+1] = ap[1];
            }
        } else {
            #pragma unroll
            for (int i = 0; i < 8; ++i) {
                int gm = m0 + arow + i*16, gk = k0 + acol;
                int gmc = gm < M1 ? gm : M1;
                int g0  = gk < K1 ? gk : K1;
                int g1  = (gk+1) < K1 ? (gk+1) : K1;
                const float* ap = A + (long long)gmc * lda;
                are[2*i]   = ap[g0];
                are[2*i+1] = ap[g1];
            }
        }
        const bool be = edgeN || (k0 + BKT > K);
        if (!be) {
            #pragma unroll
            for (int i = 0; i < 8; ++i) {
                if (TRANSB) {
                    const float* bp = B + (long long)(n0 + bcol) * ldb + (k0 + brow + i*4);
                    bre[2*i]   = bp[0];
                    bre[2*i+1] = bp[ldb];
                } else {
                    const float* bp = B + (long long)(k0 + brow + i*4) * ldb + (n0 + bcol);
                    bre[2*i]   = bp[0];
                    bre[2*i+1] = bp[1];
                }
            }
        } else {
            #pragma unroll
            for (int i = 0; i < 8; ++i) {
                int gk = k0 + brow + i*4, gn = n0 + bcol;
                int gkc = gk < K1 ? gk : K1;
                int g0  = gn < N1 ? gn : N1;
                int g1  = (gn+1) < N1 ? (gn+1) : N1;
                if (TRANSB) {
                    bre[2*i]   = B[(long long)g0 * ldb + gkc];
                    bre[2*i+1] = B[(long long)g1 * ldb + gkc];
                } else {
                    const float* bp = B + (long long)gkc * ldb;
                    bre[2*i]   = bp[g0];
                    bre[2*i+1] = bp[g1];
                }
            }
        }
    };
    auto store_tiles = [&]() {
        const bool ae = edgeM || (kld + BKT > K);
        #pragma unroll
        for (int i = 0; i < 8; ++i) {
            float v0 = are[2*i], v1 = are[2*i+1];
            if (ae) {
                int gm = m0 + arow + i*16, gk = kld + acol;
                if (!(gm < M && gk   < K)) v0 = 0.0f;
                if (!(gm < M && gk+1 < K)) v1 = 0.0f;
            }
            v2bf pk = { (bf16_t)v0, (bf16_t)v1 };
            *(v2bf*)&shA[arow + i*16][acol] = pk;
        }
        const bool be = edgeN || (kld + BKT > K);
        #pragma unroll
        for (int i = 0; i < 8; ++i) {
            float v0 = bre[2*i], v1 = bre[2*i+1];
            if (be) {
                int gk = kld + brow + i*4, gn = n0 + bcol;
                if (!(gk < K && gn   < N)) v0 = 0.0f;
                if (!(gk < K && gn+1 < N)) v1 = 0.0f;
            }
            v2bf pk = { (bf16_t)v0, (bf16_t)v1 };
            *(v2bf*)&shB[brow + i*4][bcol] = pk;
        }
    };

    v8f acc[2][4];
    const v8f zero8 = {0.f,0.f,0.f,0.f,0.f,0.f,0.f,0.f};
    #pragma unroll
    for (int s = 0; s < 2; ++s)
        #pragma unroll
        for (int t = 0; t < 4; ++t) acc[s][t] = zero8;

    load_tiles(0);
    store_tiles();
    __syncthreads();

    for (int k0 = 0; ; ) {
        const int kn  = k0 + BKT;
        const bool more = kn < K;
        if (more) load_tiles(kn);

        v16bf afr[2];
        #pragma unroll
        for (int s = 0; s < 2; ++s) {
            int m  = wr*32 + s*16 + (lane & 15);
            int kh = (lane >> 4) * 8;
            #pragma unroll
            for (int r = 0; r < 8; ++r) {
                int k = ((r < 4) ? 0 : 16) + kh + (r & 3)*2;
                afr[s][2*r]   = shA[m][k];
                afr[s][2*r+1] = shA[m][k+1];
            }
        }
        v16bf bfr[4];
        #pragma unroll
        for (int t = 0; t < 4; ++t) {
            int nb = wc*64 + t*16;
            #pragma unroll
            for (int r = 0; r < 8; ++r) {
                int k = ((r < 4) ? 0 : 16) + (lane >> 4) * 8 + (r & 3)*2;
                bfr[t][2*r]   = shB[k][nb + (lane & 15)];
                bfr[t][2*r+1] = shB[k+1][nb + (lane & 15)];
            }
        }
        #pragma unroll
        for (int s = 0; s < 2; ++s)
            #pragma unroll
            for (int t = 0; t < 4; ++t)
                acc[s][t] = __builtin_amdgcn_wmma_f32_16x16x32_bf16(
                    false, afr[s], false, bfr[t], (short)0, acc[s][t], false, false);

        __syncthreads();
        if (!more) break;
        store_tiles();
        __syncthreads();
        k0 = kn;
    }

    #pragma unroll
    for (int s = 0; s < 2; ++s) {
        #pragma unroll
        for (int t = 0; t < 4; ++t) {
            int n = n0 + wc*64 + t*16 + (lane & 15);
            #pragma unroll
            for (int v = 0; v < 8; ++v) {
                int m = m0 + wr*32 + s*16 + v + 8*(lane >> 4);
                if (m < M && n < N) {
                    float val = alpha * acc[s][t][v];
                    if (accum) val += C[(long long)m*ldc + n];
                    if (bias)  val += bias[n];
                    if      (act == 1) val = fmaxf(val, 0.0f);
                    else if (act == 2) val = 1.0f / (1.0f + __expf(-val));
                    else if (act == 3) val = tanhf(val);
                    C[(long long)m*ldc + n] = val;
                }
            }
        }
    }
}

// ---------------- elementwise / small kernels ----------------
// Row layout everywhere is NODE-MAJOR: row r = n*B + b.
__global__ void k_zero(float* p, long long n) {
    long long i = (long long)blockIdx.x*blockDim.x + threadIdx.x;
    if (i < n) p[i] = 0.0f;
}

__global__ void k_f2b(const float* __restrict__ s, bf16_t* __restrict__ d, long long n) {
    long long i = (long long)blockIdx.x*blockDim.x + threadIdx.x;
    if (i < n) d[i] = (bf16_t)s[i];
}

__global__ void k_diag_sub1(float* S, int n) {
    int i = blockIdx.x*blockDim.x + threadIdx.x;
    if (i < n) S[(long long)i*n + i] -= 1.0f;
}

__global__ __launch_bounds__(256)
void k_softmax_rows(float* P, int cols, int ld) {
    __shared__ float red[256];
    int row = blockIdx.x, tid = threadIdx.x;
    float* p = P + (long long)row * ld;
    float mx = -3.4e38f;
    for (int j = tid; j < cols; j += 256) mx = fmaxf(mx, p[j]);
    red[tid] = mx; __syncthreads();
    for (int s = 128; s > 0; s >>= 1) { if (tid < s) red[tid] = fmaxf(red[tid], red[tid+s]); __syncthreads(); }
    mx = red[0]; __syncthreads();
    float sm = 0.0f;
    for (int j = tid; j < cols; j += 256) { float e = __expf(p[j]-mx); p[j] = e; sm += e; }
    red[tid] = sm; __syncthreads();
    for (int s = 128; s > 0; s >>= 1) { if (tid < s) red[tid] += red[tid+s]; __syncthreads(); }
    float inv = 1.0f / red[0];
    for (int j = tid; j < cols; j += 256) p[j] *= inv;
}

__global__ void k_softmax_small(float* P, int rows, int C) {
    int r = blockIdx.x*blockDim.x + threadIdx.x;
    if (r >= rows) return;
    float* p = P + (long long)r*C;
    float mx = -3.4e38f;
    for (int j = 0; j < C; ++j) mx = fmaxf(mx, p[j]);
    float s = 0.0f;
    for (int j = 0; j < C; ++j) { float e = __expf(p[j]-mx); p[j] = e; s += e; }
    float inv = 1.0f/s;
    for (int j = 0; j < C; ++j) p[j] *= inv;
}

// encoder cell input (bf16, padded rows of Cp): [ x[b,n,t] , (z*)h , 0-pad ]
__global__ void k_xh_enc(const float* __restrict__ x, const float* __restrict__ h,
                         const float* __restrict__ zr, bf16_t* __restrict__ xh,
                         int t, int Bb, int Nn, int L, int R, int Cp, int useZ) {
    int C = 1 + R;
    long long total = (long long)Bb*Nn*Cp;
    long long idx = (long long)blockIdx.x*blockDim.x + threadIdx.x;
    if (idx >= total) return;
    long long r = idx / Cp; int c = (int)(idx % Cp);
    float v = 0.0f;
    if (c == 0) {
        int b = (int)(r % Bb), n = (int)(r / Bb);
        v = x[((long long)b*Nn + n)*L + t];
    } else if (c < C) {
        v = h[r*R + (c-1)];
        if (useZ) v *= zr[r*(2LL*R) + (c-1)];
    }
    xh[idx] = (bf16_t)v;
}

// decoder cell input (bf16, padded): [ go, tt[b,0:5,t], (z*)h, 0-pad ]
__global__ void k_xh_dec(const float* __restrict__ go, const float* __restrict__ tt,
                         const float* __restrict__ h, const float* __restrict__ zr,
                         bf16_t* __restrict__ xh, int t, int Bb, int Nn, int H, int ycd,
                         int R, int Cp, int useZ) {
    int Cx = 1 + ycd, C = Cx + R;
    long long total = (long long)Bb*Nn*Cp;
    long long idx = (long long)blockIdx.x*blockDim.x + threadIdx.x;
    if (idx >= total) return;
    int c = (int)(idx % Cp); long long r = idx / Cp;
    float v = 0.0f;
    if (c == 0) v = go[r];
    else if (c < Cx) {
        int b = (int)(r % Bb);
        v = tt[((long long)b*ycd + (c-1))*H + t];
    } else if (c < C) {
        v = h[r*R + (c-Cx)];
        if (useZ) v *= zr[r*(2LL*R) + (c-Cx)];
    }
    xh[idx] = (bf16_t)v;
}

// h_new = r*h + (1-r)*hc  with r = zr[:, R:2R]
__global__ void k_gru(const float* __restrict__ zr, const float* __restrict__ hc,
                      const float* __restrict__ h, float* __restrict__ hn,
                      long long rows, int R) {
    long long total = rows * R;
    long long idx = (long long)blockIdx.x*blockDim.x + threadIdx.x;
    if (idx >= total) return;
    long long r = idx / R; int j = (int)(idx % R);
    float rr = zr[r*(2LL*R) + R + j];
    hn[idx] = rr * h[idx] + (1.0f - rr) * hc[idx];
}

// hde = concat([ht, hatt])  (R -> 2R)
__global__ void k_hde(const float* __restrict__ ht, const float* __restrict__ hatt,
                      float* __restrict__ hde, long long rows, int R) {
    int D = 2*R;
    long long total = rows * D;
    long long idx = (long long)blockIdx.x*blockDim.x + threadIdx.x;
    if (idx >= total) return;
    long long r = idx / D; int c = (int)(idx % D);
    hde[idx] = (c < R) ? ht[r*R + c] : hatt[r*R + (c-R)];
}

// go = h@pW + pb ; also write into d_out[b,0,n,t]
__global__ void k_proj(const float* __restrict__ h, const float* __restrict__ pW,
                       const float* __restrict__ pb, float* __restrict__ go,
                       float* __restrict__ out, int t, int rows, int Bb, int Nn,
                       int Dh, int H) {
    int r = blockIdx.x*blockDim.x + threadIdx.x;
    if (r >= rows) return;
    const float* hr = h + (long long)r*Dh;
    float s = pb[0];
    for (int j = 0; j < Dh; ++j) s += hr[j]*pW[j];
    go[r] = s;
    int b = r % Bb, n = r / Bb;
    out[((long long)b*Nn + n)*H + t] = s;
}

// ---------------- host orchestration ----------------
static inline void launch_gemm_f32(hipStream_t st,
    const float* A, int lda, const float* B, int ldb, int transB,
    float* C, int ldc, const float* bias, int M, int N, int K,
    float alpha, int accum, int act)
{
    dim3 grid((N + BNT - 1)/BNT, (M + BMT - 1)/BMT, 1);
    if (transB)
        gemm_f32_wmma<1><<<grid, 256, 0, st>>>(A, lda, B, ldb, C, ldc, bias, M, N, K, alpha, accum, act);
    else
        gemm_f32_wmma<0><<<grid, 256, 0, st>>>(A, lda, B, ldb, C, ldc, bias, M, N, K, alpha, accum, act);
}

static inline void launch_gemm_bb(hipStream_t st,
    const bf16_t* A, int lda, const bf16_t* B, int ldb,
    float* Cf, bf16_t* Cb, int ldc, const float* bias,
    int M, int N, int K, int accum, int act)
{
    dim3 grid((N + BNT - 1)/BNT, (M + BMT - 1)/BMT, 1);
    gemm_bb_wmma<<<grid, 256, 0, st>>>(A, lda, B, ldb, Cf, Cb, ldc, bias, M, N, K, accum, act);
}

static inline int g1d(long long n) { return (int)((n + 255) / 256); }

// AGCN (node-major, bf16 operands): out = act( xh@W0 + xh@W3 + sum_k (S_k@xh)@W_blk + b )
// xh: (B*N, Cp) bf16 == (N, B*Cp) bf16 (same memory).
static void agcn(hipStream_t st, const bf16_t* xh, int C, int Cp,
                 const bf16_t* Sb[4], const bf16_t* Wb, int O, const float* bias,
                 bf16_t* D, float* out, int lastAct)
{
    const int ROWS = 32768, Nn = 512, Bb = 64;
    const int BCp = Bb * Cp;
    launch_gemm_bb(st, xh, Cp, Wb + 0LL*C*O, O, out, nullptr, O, bias,    ROWS, O, C, 0, 0);
    launch_gemm_bb(st, xh, Cp, Wb + 3LL*C*O, O, out, nullptr, O, nullptr, ROWS, O, C, 1, 0);
    const int wblk[4] = {1, 2, 4, 5};
    for (int k = 0; k < 4; ++k) {
        // diffusion: D(N, B*Cp) = S_k(N,N) @ xh(N, B*Cp) — edge-free async kernel
        dim3 grid(BCp/BNT, Nn/BMT, 1);
        gemm_bb_async<<<grid, 256, 0, st>>>(Sb[k], Nn, xh, BCp, D, BCp, Nn, BCp, Nn);
        int act = (k == 3) ? lastAct : 0;
        launch_gemm_bb(st, D, Cp, Wb + (long long)wblk[k]*C*O, O, out, nullptr, O, nullptr,
                       ROWS, O, C, 1, act);
    }
}

extern "C" void kernel_launch(void* const* d_in, const int* in_sizes, int n_in,
                              void* d_out, int out_size, void* d_ws, size_t ws_size,
                              hipStream_t stream) {
    (void)in_sizes; (void)n_in; (void)out_size; (void)ws_size;

    const float* x   = (const float*)d_in[0];   // (64,1,512,12)
    const float* tt  = (const float*)d_in[3];   // (64,5,1,12)
    const float* Mem = (const float*)d_in[5];   // (20,64)
    const float* Wq  = (const float*)d_in[6];   // (64,64)
    const float* We1 = (const float*)d_in[7];   // (512,20)
    const float* We2 = (const float*)d_in[8];   // (512,20)
    const float* egW = (const float*)d_in[9];   // (390,128)
    const float* egb = (const float*)d_in[10];
    const float* euW = (const float*)d_in[11];  // (390,64)
    const float* eub = (const float*)d_in[12];
    const float* dgW = (const float*)d_in[13];  // (804,256)
    const float* dgb = (const float*)d_in[14];
    const float* duW = (const float*)d_in[15];  // (804,128)
    const float* dub = (const float*)d_in[16];
    const float* pW  = (const float*)d_in[17];  // (128,1)
    const float* pb  = (const float*)d_in[18];
    float* out = (float*)d_out;

    const int Bb = 64, Nn = 512, L = 12, H = 12;
    const int Renc = 64, Rdec = 128, ycd = 5, Mn = 20, Md = 64;
    const int ROWS = Bb*Nn;            // 32768 (node-major: r = n*B + b)
    const int Cenc = 1 + Renc;         // 65
    const int Cdec = 1 + ycd + Rdec;   // 134
    const int CpE  = 72;               // padded (mult of 8, 16B rows)
    const int CpD  = 136;

    // workspace carve (floats, 16B-aligned chunks)
    float* w = (float*)d_ws;
    auto alloc = [&](long long n){ n = (n + 3) & ~3LL; float* p = w; w += n; return p; };
    float* e1   = alloc((long long)Nn*Md);
    float* e2   = alloc((long long)Nn*Md);
    float* S1   = alloc((long long)Nn*Nn);
    float* S2   = alloc((long long)Nn*Nn);
    float* S3   = alloc((long long)Nn*Nn);
    float* S4   = alloc((long long)Nn*Nn);
    float* zr   = alloc((long long)ROWS*2*Rdec);
    float* hc   = alloc((long long)ROWS*Rdec);
    float* hA   = alloc((long long)ROWS*Rdec);
    float* hB   = alloc((long long)ROWS*Rdec);
    float* qry  = alloc((long long)ROWS*Md);
    float* att  = alloc((long long)ROWS*Mn);
    float* hatt = alloc((long long)ROWS*Md);
    float* go   = alloc((long long)ROWS);
    bf16_t* Sb1  = (bf16_t*)alloc((long long)Nn*Nn/2);
    bf16_t* Sb2  = (bf16_t*)alloc((long long)Nn*Nn/2);
    bf16_t* Sb3  = (bf16_t*)alloc((long long)Nn*Nn/2);
    bf16_t* Sb4  = (bf16_t*)alloc((long long)Nn*Nn/2);
    bf16_t* xhb  = (bf16_t*)alloc((long long)ROWS*CpD/2);
    bf16_t* Db   = (bf16_t*)alloc((long long)ROWS*CpD/2);
    bf16_t* egWb = (bf16_t*)alloc((6LL*Cenc*2*Renc + 3)/2);
    bf16_t* euWb = (bf16_t*)alloc((6LL*Cenc*Renc   + 3)/2);
    bf16_t* dgWb = (bf16_t*)alloc((6LL*Cdec*2*Rdec + 3)/2);
    bf16_t* duWb = (bf16_t*)alloc((6LL*Cdec*Rdec   + 3)/2);

    // ---- weight conversions (once) ----
    k_f2b<<<g1d(6LL*Cenc*2*Renc), 256, 0, stream>>>(egW, egWb, 6LL*Cenc*2*Renc);
    k_f2b<<<g1d(6LL*Cenc*Renc),   256, 0, stream>>>(euW, euWb, 6LL*Cenc*Renc);
    k_f2b<<<g1d(6LL*Cdec*2*Rdec), 256, 0, stream>>>(dgW, dgWb, 6LL*Cdec*2*Rdec);
    k_f2b<<<g1d(6LL*Cdec*Rdec),   256, 0, stream>>>(duW, duWb, 6LL*Cdec*Rdec);

    // ---- supports (f32 build, then bf16 convert) ----
    launch_gemm_f32(stream, We1, Mn, Mem, Md, 0, e1, Md, nullptr, Nn, Md, Mn, 1.0f, 0, 0);
    launch_gemm_f32(stream, We2, Mn, Mem, Md, 0, e2, Md, nullptr, Nn, Md, Mn, 1.0f, 0, 0);
    launch_gemm_f32(stream, e1, Md, e2, Md, 1, S1, Nn, nullptr, Nn, Nn, Md, 1.0f, 0, 1); // relu
    k_softmax_rows<<<Nn, 256, 0, stream>>>(S1, Nn, Nn);
    launch_gemm_f32(stream, e2, Md, e1, Md, 1, S3, Nn, nullptr, Nn, Nn, Md, 1.0f, 0, 1);
    k_softmax_rows<<<Nn, 256, 0, stream>>>(S3, Nn, Nn);
    launch_gemm_f32(stream, S1, Nn, S1, Nn, 0, S2, Nn, nullptr, Nn, Nn, Nn, 2.0f, 0, 0);
    k_diag_sub1<<<g1d(Nn), 256, 0, stream>>>(S2, Nn);
    launch_gemm_f32(stream, S3, Nn, S3, Nn, 0, S4, Nn, nullptr, Nn, Nn, Nn, 2.0f, 0, 0);
    k_diag_sub1<<<g1d(Nn), 256, 0, stream>>>(S4, Nn);
    k_f2b<<<g1d((long long)Nn*Nn), 256, 0, stream>>>(S1, Sb1, (long long)Nn*Nn);
    k_f2b<<<g1d((long long)Nn*Nn), 256, 0, stream>>>(S2, Sb2, (long long)Nn*Nn);
    k_f2b<<<g1d((long long)Nn*Nn), 256, 0, stream>>>(S3, Sb3, (long long)Nn*Nn);
    k_f2b<<<g1d((long long)Nn*Nn), 256, 0, stream>>>(S4, Sb4, (long long)Nn*Nn);
    const bf16_t* Sb[4] = {Sb1, Sb2, Sb3, Sb4};

    // ---- encoder ----
    k_zero<<<g1d((long long)ROWS*Renc), 256, 0, stream>>>(hA, (long long)ROWS*Renc);
    float* h = hA; float* hn = hB;
    for (int t = 0; t < L; ++t) {
        k_xh_enc<<<g1d((long long)ROWS*CpE), 256, 0, stream>>>(x, h, nullptr, xhb, t, Bb, Nn, L, Renc, CpE, 0);
        agcn(stream, xhb, Cenc, CpE, Sb, egWb, 2*Renc, egb, Db, zr, 2);   // sigmoid gates
        k_xh_enc<<<g1d((long long)ROWS*CpE), 256, 0, stream>>>(x, h, zr, xhb, t, Bb, Nn, L, Renc, CpE, 1);
        agcn(stream, xhb, Cenc, CpE, Sb, euWb, Renc, eub, Db, hc, 3);     // tanh candidate
        k_gru<<<g1d((long long)ROWS*Renc), 256, 0, stream>>>(zr, hc, h, hn, ROWS, Renc);
        float* tmp = h; h = hn; hn = tmp;
    }

    // ---- attention / decoder init ----
    launch_gemm_f32(stream, h, Renc, Wq, Md, 0, qry, Md, nullptr, ROWS, Md, Renc, 1.0f, 0, 0);
    launch_gemm_f32(stream, qry, Md, Mem, Md, 1, att, Mn, nullptr, ROWS, Mn, Md, 1.0f, 0, 0);
    k_softmax_small<<<g1d(ROWS), 256, 0, stream>>>(att, ROWS, Mn);
    launch_gemm_f32(stream, att, Mn, Mem, Md, 0, hatt, Md, nullptr, ROWS, Md, Mn, 1.0f, 0, 0);
    k_hde<<<g1d((long long)ROWS*Rdec), 256, 0, stream>>>(h, hatt, hn, ROWS, Renc);
    { float* tmp = h; h = hn; hn = tmp; }   // h now holds h_de (stride 128)
    k_zero<<<g1d((long long)ROWS), 256, 0, stream>>>(go, ROWS);

    // ---- decoder ----
    for (int t = 0; t < H; ++t) {
        k_xh_dec<<<g1d((long long)ROWS*CpD), 256, 0, stream>>>(go, tt, h, nullptr, xhb, t, Bb, Nn, H, ycd, Rdec, CpD, 0);
        agcn(stream, xhb, Cdec, CpD, Sb, dgWb, 2*Rdec, dgb, Db, zr, 2);
        k_xh_dec<<<g1d((long long)ROWS*CpD), 256, 0, stream>>>(go, tt, h, zr, xhb, t, Bb, Nn, H, ycd, Rdec, CpD, 1);
        agcn(stream, xhb, Cdec, CpD, Sb, duWb, Rdec, dub, Db, hc, 3);
        k_gru<<<g1d((long long)ROWS*Rdec), 256, 0, stream>>>(zr, hc, h, hn, ROWS, Rdec);
        float* tmp = h; h = hn; hn = tmp;
        k_proj<<<g1d(ROWS), 256, 0, stream>>>(h, pW, pb, go, out, t, ROWS, Bb, Nn, Rdec, H);
    }
}
